// InfoNCELoss_86620900426225
// MI455X (gfx1250) — matrix operand order repaired and checked by reading, not compiled, not run on hardware
//
#include <hip/hip_runtime.h>
#include <hip/hip_bf16.h>

typedef __attribute__((ext_vector_type(16))) _Float16 v16h;
typedef __attribute__((ext_vector_type(8)))  _Float16 v8h;
typedef __attribute__((ext_vector_type(8)))  float    v8f;

#define TEMP_INV 10.0f   // 1 / TEMPERATURE

// ---------------------------------------------------------------------------
// Kernel 1: L2-normalize rows (fp32 math), store f16; zero per-row pos/den.
// One 128-thread block per row.
// ---------------------------------------------------------------------------
__global__ void nrm_kernel(const float* __restrict__ feats,
                           _Float16* __restrict__ fnorm,
                           float* __restrict__ posArr,
                           float* __restrict__ denArr,
                           int D) {
    const int row = blockIdx.x;
    const int tid = threadIdx.x;
    if (tid == 0) { posArr[row] = 0.0f; denArr[row] = 0.0f; }

    const float* fr = feats + (size_t)row * D;
    float s = 0.0f;
    for (int k = tid; k < D; k += blockDim.x) {
        float v = fr[k];
        s += v * v;
    }
    __shared__ float red[128];
    red[tid] = s;
    __syncthreads();
    for (int off = 64; off > 0; off >>= 1) {
        if (tid < off) red[tid] += red[tid + off];
        __syncthreads();
    }
    const float inv = 1.0f / fmaxf(sqrtf(red[0]), 1e-12f);
    _Float16* fo = fnorm + (size_t)row * D;
    for (int k = tid; k < D; k += blockDim.x)
        fo[k] = (_Float16)(fr[k] * inv);
}

// ---------------------------------------------------------------------------
// Kernel 2: fused sim-tile + masked exp-sum.
// One wave32 per 64x64 output tile, 4x4 blocking of 16x16x32 f16 WMMA.
// ---------------------------------------------------------------------------
__global__ void __launch_bounds__(32)
sim_kernel(const _Float16* __restrict__ fnorm,
           const int* __restrict__ labels,
           float* __restrict__ posArr,
           float* __restrict__ denArr,
           int D) {
    const int lane    = threadIdx.x;        // 0..31 (wave32)
    const int l15     = lane & 15;
    const int hi      = lane >> 4;          // 0 or 1
    const int rowBase = blockIdx.y * 64;
    const int colBase = blockIdx.x * 64;

    v8f acc[4][4];
#pragma unroll
    for (int ti = 0; ti < 4; ++ti)
#pragma unroll
        for (int tj = 0; tj < 4; ++tj)
            acc[ti][tj] = (v8f){0.f, 0.f, 0.f, 0.f, 0.f, 0.f, 0.f, 0.f};

    // Per-lane base pointers.
    // A frag (16x32, MxK): lane l<16 -> row M=l, K=0..15 contiguous;
    //                      lane l+16 -> row M=l, K=16..31 contiguous.
    // B frag (32x16, KxN): lane l<16 -> col N=l, K=0..15 contiguous;
    //                      lane l+16 -> col N=l, K=16..31 contiguous.
    // B = f^T, so column n of B is feature row n: identical storage, no transpose.
    const _Float16* aRow[4];
    const _Float16* bCol[4];
#pragma unroll
    for (int t = 0; t < 4; ++t) {
        aRow[t] = fnorm + (size_t)(rowBase + t * 16 + l15) * D + hi * 16;
        bCol[t] = fnorm + (size_t)(colBase + t * 16 + l15) * D + hi * 16;
    }

    for (int k0 = 0; k0 < D; k0 += 32) {
        v16h a[4], b[4];
#pragma unroll
        for (int t = 0; t < 4; ++t) {
            v8h lo  = *(const v8h*)(aRow[t] + k0);        // K = k0 + hi*16 + 0..7
            v8h hi8 = *(const v8h*)(aRow[t] + k0 + 8);    // K = k0 + hi*16 + 8..15
            a[t] = __builtin_shufflevector(lo, hi8,
                    0, 1, 2, 3, 4, 5, 6, 7, 8, 9, 10, 11, 12, 13, 14, 15);
            b[t] = *(const v16h*)(bCol[t] + k0);          // K = k0 + hi*16 + 0..15
        }
#pragma unroll
        for (int ti = 0; ti < 4; ++ti)
#pragma unroll
            for (int tj = 0; tj < 4; ++tj)
                acc[ti][tj] = __builtin_amdgcn_wmma_f32_16x16x32_f16(
                    /*neg_a=*/false, a[ti],
                    /*neg_b=*/false, b[tj],
                    /*c_mod=*/(short)0, acc[ti][tj],
                    /*reuse_a=*/false, /*reuse_b=*/false);
    }

    // Epilogue: C/D layout -> element (vgpr v, lane): row = base+ti*16+v+8*hi,
    // col = base+tj*16+(lane&15). exp, mask, 16-lane reduce, fp32 atomics.
    int labJ[4];
#pragma unroll
    for (int tj = 0; tj < 4; ++tj)
        labJ[tj] = labels[colBase + tj * 16 + l15];

#pragma unroll
    for (int ti = 0; ti < 4; ++ti) {
#pragma unroll
        for (int v = 0; v < 8; ++v) {
            const int i    = rowBase + ti * 16 + v + hi * 8;
            const int labI = labels[i];
            float den = 0.0f, pos = 0.0f;
#pragma unroll
            for (int tj = 0; tj < 4; ++tj) {
                const int j  = colBase + tj * 16 + l15;
                const float e = __expf(acc[ti][tj][v] * TEMP_INV);
                const bool offd = (i != j);
                den += offd ? e : 0.0f;
                pos += (offd && (labI == labJ[tj])) ? e : 0.0f;
            }
            // Reduce across the 16 lanes sharing row i (xor masks < 16 stay
            // inside each half-wave group).
#pragma unroll
            for (int m = 1; m < 16; m <<= 1) {
                den += __shfl_xor(den, m, 32);
                pos += __shfl_xor(pos, m, 32);
            }
            if (l15 == 0) {
                atomicAdd(&denArr[i], den);
                atomicAdd(&posArr[i], pos);
            }
        }
    }
}

// ---------------------------------------------------------------------------
// Kernel 3: loss_i = log(den) - log(pos) over rows with positives; mean.
// ---------------------------------------------------------------------------
__global__ void fin_kernel(const float* __restrict__ posArr,
                           const float* __restrict__ denArr,
                           float* __restrict__ out, int N) {
    const int tid = threadIdx.x;
    float sum = 0.0f;
    int   cnt = 0;
    for (int i = tid; i < N; i += 256) {
        const float p = posArr[i];
        if (p > 0.0f) {
            sum += logf(denArr[i]) - logf(p);
            cnt++;
        }
    }
    __shared__ float sSum[256];
    __shared__ int   sCnt[256];
    sSum[tid] = sum;
    sCnt[tid] = cnt;
    __syncthreads();
    for (int off = 128; off > 0; off >>= 1) {
        if (tid < off) { sSum[tid] += sSum[tid + off]; sCnt[tid] += sCnt[tid + off]; }
        __syncthreads();
    }
    if (tid == 0)
        out[0] = (sCnt[0] > 0) ? (sSum[0] / (float)sCnt[0]) : 0.0f;
}

// ---------------------------------------------------------------------------
extern "C" void kernel_launch(void* const* d_in, const int* in_sizes, int n_in,
                              void* d_out, int out_size, void* d_ws, size_t ws_size,
                              hipStream_t stream) {
    const float* feats  = (const float*)d_in[0];
    const int*   labels = (const int*)d_in[1];
    float*       out    = (float*)d_out;

    const int N = in_sizes[1];              // 8192
    const int D = in_sizes[0] / N;          // 512

    // Workspace: [f16 normalized feats | posArr | denArr]
    _Float16* fnorm  = (_Float16*)d_ws;
    float*    posArr = (float*)((char*)d_ws + (size_t)N * D * sizeof(_Float16));
    float*    denArr = posArr + N;

    nrm_kernel<<<N, 128, 0, stream>>>(feats, fnorm, posArr, denArr, D);

    dim3 grid(N / 64, N / 64);
    sim_kernel<<<grid, 32, 0, stream>>>(fnorm, labels, posArr, denArr, D);

    fin_kernel<<<1, 256, 0, stream>>>(posArr, denArr, out, N);
}
